// FlashRotarySelfAttention_51874615001214
// MI455X (gfx1250) — matrix operand order repaired and checked by reading, not compiled, not run on hardware
//
#include <hip/hip_runtime.h>

// Problem constants (B=2, S=2048, C=2048, H=16, Dh=128)
#define B_    2
#define S_    2048
#define C_    2048
#define H_    16
#define DH_   128
#define NTOK  (B_ * S_)      // 4096 tokens
#define QKVC  (3 * C_)       // 6144

typedef __attribute__((ext_vector_type(16))) _Float16     v16h;
typedef __attribute__((ext_vector_type(8)))  float        v8f;
typedef __attribute__((ext_vector_type(4)))  unsigned int v4u;

// Matches the builtin's parameter type exactly (per hipcc diagnostic):
// 'int __attribute__((vector_size(16)))'.
typedef int v4i_ __attribute__((vector_size(16)));

// One WMMA operand fragment: 16 halves per lane = 8 VGPRs = two b128 loads.
union Frag { v16h h; v4u u[2]; };

static __device__ inline v8f wmma16x16x32(const Frag& a, const Frag& b, v8f c) {
  // D = A(16x32 f16) * B(32x16 f16) + C(16x16 f32)
  return __builtin_amdgcn_wmma_f32_16x16x32_f16(
      /*neg_a=*/false, a.h, /*neg_b=*/false, b.h,
      /*c_mod=*/(short)0, c, /*reuse_a=*/false, /*reuse_b=*/false);
}

// ---------------------------------------------------------------------------
// CDNA5 async global->LDS copy (GLOBAL_LOAD_ASYNC_TO_LDS_B128, ASYNCcnt).
// Builtin confirmed present in this toolchain (round-2 diagnostic).
// ---------------------------------------------------------------------------
#if defined(__has_builtin)
#if __has_builtin(__builtin_amdgcn_global_load_async_to_lds_b128)
#define ASYNC_LDS_BUILTIN 1
#endif
#if __has_builtin(__builtin_amdgcn_s_wait_asynccnt)
#define ASYNC_WAIT_BUILTIN 1
#endif
#endif

static __device__ inline void async_copy16(void* lds_dst, const void* gsrc) {
#if defined(ASYNC_LDS_BUILTIN)
  __builtin_amdgcn_global_load_async_to_lds_b128(
      (__attribute__((address_space(1))) v4i_*)const_cast<void*>(gsrc),
      (__attribute__((address_space(3))) v4i_*)lds_dst,
      /*imm offset=*/0, /*cpol=*/0);
#else
  // vdst = 32-bit LDS byte address (low 32 bits of the generic LDS pointer),
  // vaddr = 64-bit global address, GV mode.
  asm volatile("global_load_async_to_lds_b128 %0, %1, off"
               :
               : "v"((unsigned)(unsigned long long)(size_t)lds_dst),
                 "v"((unsigned long long)(size_t)gsrc)
               : "memory");
#endif
}

static __device__ inline void wait_async0() {
#if defined(ASYNC_WAIT_BUILTIN)
  __builtin_amdgcn_s_wait_asynccnt(0);
#else
  asm volatile("s_wait_asynccnt 0x0" ::: "memory");
#endif
}

// ---------------------------------------------------------------------------
// Kernel 1: fp32 -> f16 cast (weights / activations enter the WMMA domain)
// ---------------------------------------------------------------------------
__global__ void cast_f32_to_f16(const float* __restrict__ src,
                                _Float16* __restrict__ dst, int n) {
  int i = blockIdx.x * blockDim.x + threadIdx.x;
  if (i < n) dst[i] = (_Float16)src[i];
}

// ---------------------------------------------------------------------------
// Kernel 2/5: f16 WMMA GEMM, C[MxN] = A[MxK] * B[KxN], fp32 accumulate.
// Block tile 128x256, 256 threads = 8 waves, wave tile 32x128 (2x8 WMMA
// tiles -> 16 WMMA per K-step vs 10 fragment loads).
// A tile is staged into LDS with async global->LDS b128 copies (ASYNCcnt);
// B tile is transposed through VGPRs into [n][k] so a B fragment is two
// contiguous ds_load_b128 per lane. LDS is double-buffered: tile t+1 streams
// in while WMMAs consume tile t; one s_wait_asynccnt + one barrier per step.
// ---------------------------------------------------------------------------
template <typename TOUT>
__global__ void __launch_bounds__(256)
gemm_f16_wmma(const _Float16* __restrict__ A, const _Float16* __restrict__ Bm,
              TOUT* __restrict__ Cm, int M, int N, int K) {
  __shared__ _Float16 ldsA[2][128][32];    // 2 x 8 KB
  __shared__ _Float16 ldsBT[2][256][32];   // 2 x 16 KB  (transposed [n][k])

  const int tid  = threadIdx.x;
  const int wave = tid >> 5, lane = tid & 31;
  const int l16  = lane & 15, hi = lane >> 4;   // hi: which 16-lane half
  const int m_blk = blockIdx.y * 128;
  const int n_blk = blockIdx.x * 256;
  const int wm = (wave & 3) * 32;               // 4 waves tile M
  const int wn = (wave >> 2) * 128;             // 2 waves tile N

  v8f acc[2][8] = {};

  auto stage = [&](int kbase, int buf) {
    // A tile 128x32: 512 x 16B chunks, async straight into LDS.
#pragma unroll
    for (int c = 0; c < 2; ++c) {
      int idx = tid + c * 256;
      int r = idx >> 2, kk = (idx & 3) << 3;
      async_copy16(&ldsA[buf][r][kk],
                   A + (size_t)(m_blk + r) * K + (kbase + kk));
    }
    // B tile 32x256 -> transposed LDS [n][k] (transpose goes through VGPRs).
#pragma unroll
    for (int c = 0; c < 4; ++c) {
      int idx = tid + c * 256;
      int kk = idx >> 5, n0 = (idx & 31) << 3;
      const _Float16* bp = Bm + (size_t)(kbase + kk) * N + (n_blk + n0);
      v4u t = *(const v4u*)bp;
      if (kbase + 32 < K)
        __builtin_prefetch(bp + (size_t)32 * N, 0, 1);  // next K-tile row
      union { v4u v; unsigned short s[8]; } u; u.v = t;
      union { unsigned short s; _Float16 hh; } cv;
#pragma unroll
      for (int i = 0; i < 8; ++i) { cv.s = u.s[i]; ldsBT[buf][n0 + i][kk] = cv.hh; }
    }
  };

  stage(0, 0);
  int cur = 0;
  for (int k0 = 0; k0 < K; k0 += 32) {
    wait_async0();        // our async A chunks for `cur` have landed
    __syncthreads();      // everyone's staging of `cur` is visible
    if (k0 + 32 < K) stage(k0 + 32, cur ^ 1);  // stream next tile

    // A fragment: lane l16 = row; half picks k {hi*8..+7, 16+hi*8..+7}.
    Frag a[2], b[8];
#pragma unroll
    for (int i = 0; i < 2; ++i) {
      const _Float16* p = &ldsA[cur][wm + i * 16 + l16][0];
      a[i].u[0] = *(const v4u*)(p + hi * 8);
      a[i].u[1] = *(const v4u*)(p + 16 + hi * 8);
    }
    // B fragment: lane l16 = col; half picks k {hi*8..+7, 16+hi*8..+7}.
#pragma unroll
    for (int j = 0; j < 8; ++j) {
      const _Float16* p = &ldsBT[cur][wn + j * 16 + l16][0];
      b[j].u[0] = *(const v4u*)(p + hi * 8);
      b[j].u[1] = *(const v4u*)(p + 16 + hi * 8);
    }
#pragma unroll
    for (int i = 0; i < 2; ++i)
#pragma unroll
      for (int j = 0; j < 8; ++j)
        acc[i][j] = wmma16x16x32(a[i], b[j], acc[i][j]);
    cur ^= 1;
  }

  // C layout: VGPR e -> row hi*8+e (lanes 0-15: M=e, lanes 16-31: M=8+e).
#pragma unroll
  for (int i = 0; i < 2; ++i)
#pragma unroll
    for (int j = 0; j < 8; ++j) {
      int col = n_blk + wn + j * 16 + l16;
#pragma unroll
      for (int e = 0; e < 8; ++e) {
        int row = m_blk + wm + i * 16 + hi * 8 + e;
        Cm[(size_t)row * N + col] = (TOUT)acc[i][j][e];
      }
    }
}

// ---------------------------------------------------------------------------
// Kernel 3: in-place RoPE on the k (cols 0..C) and q (cols C..2C) thirds.
// One thread per (token, part, head, rotation pair).
// ---------------------------------------------------------------------------
__global__ void rope_inplace(_Float16* __restrict__ qkv) {
  int idx = blockIdx.x * blockDim.x + threadIdx.x;
  int i   = idx & 63;           // rotation pair 0..63 (half = Dh/2)
  int h   = (idx >> 6) & 15;    // head
  int prt = (idx >> 10) & 1;    // 0 = k, 1 = q
  int t   = idx >> 11;          // token
  if (t >= NTOK) return;
  int pos = t & (S_ - 1);       // sequence position
  float invf = __powf(10000.0f, -(float)i * (1.0f / 64.0f));
  float ang = (float)pos * invf;
  float sn, cs;
  __sincosf(ang, &sn, &cs);
  size_t base = (size_t)t * QKVC + (size_t)prt * C_ + (size_t)h * DH_;
  float t1 = (float)qkv[base + i];
  float t2 = (float)qkv[base + 64 + i];
  qkv[base + i]      = (_Float16)(t1 * cs - t2 * sn);
  qkv[base + 64 + i] = (_Float16)(t2 * cs + t1 * sn);
}

// ---------------------------------------------------------------------------
// Kernel 4: causal flash attention, WMMA for QK^T and PV.
// Block = 128 threads (4 waves); each wave owns 16 query rows of a 64-row
// q-tile. Q/K fragments come straight from global (their fragment layouts
// are contiguous per token row); V is transposed through LDS for the PV
// matmul; probabilities are restaged through LDS (C-layout -> A-layout).
// ---------------------------------------------------------------------------
__global__ void __launch_bounds__(128)
flash_attn_wmma(const _Float16* __restrict__ qkv, _Float16* __restrict__ aout) {
  __shared__ _Float16 ldsVT[128][64];   // V transposed: [d][kv]
  __shared__ _Float16 ldsP[64][64];     // probabilities: [q][kv]

  const int tid  = threadIdx.x;
  const int wave = tid >> 5, lane = tid & 31;
  const int l16  = lane & 15, hi = lane >> 4;
  const int qt = blockIdx.x & 31;       // q tile (S/64 = 32)
  const int bh = blockIdx.x >> 5;
  const int b  = bh >> 4, h = bh & 15;
  const size_t rs   = QKVC;
  const size_t tok0 = (size_t)b * S_;
  const _Float16* Kp = qkv + tok0 * rs + 0      + (size_t)h * DH_;  // split: k,q,v
  const _Float16* Qp = qkv + tok0 * rs + C_     + (size_t)h * DH_;
  const _Float16* Vp = qkv + tok0 * rs + 2 * C_ + (size_t)h * DH_;
  const int q0 = qt * 64;
  const float scale = 0.088388347648318447f;    // Dh^-0.5

  // Preload Q fragments for the wave's 16 rows (4 k-steps of 32 over Dh=128).
  Frag aq[4];
  {
    const _Float16* qr = Qp + (size_t)(q0 + wave * 16 + l16) * rs;
#pragma unroll
    for (int d = 0; d < 4; ++d) {
      aq[d].u[0] = *(const v4u*)(qr + d * 32 + hi * 8);
      aq[d].u[1] = *(const v4u*)(qr + d * 32 + 16 + hi * 8);
    }
  }

  v8f acc[8] = {};                      // O accumulator: 16 x 128
  float mrow[8], lrow[8];
#pragma unroll
  for (int e = 0; e < 8; ++e) { mrow[e] = -1e30f; lrow[e] = 0.0f; }

  for (int kt = 0; kt <= qt; ++kt) {    // causal: only tiles at/below diagonal
    const int k0 = kt * 64;

    // Cooperative transpose-load of V tile 64x128 -> ldsVT[d][kv].
#pragma unroll
    for (int c = 0; c < 8; ++c) {
      int idx = tid + c * 128;
      int r = idx >> 4, d0 = (idx & 15) << 3;
      v4u t = *(const v4u*)(Vp + (size_t)(k0 + r) * rs + d0);
      union { v4u v; unsigned short s[8]; } u; u.v = t;
      union { unsigned short s; _Float16 hh; } cv;
#pragma unroll
      for (int i = 0; i < 8; ++i) { cv.s = u.s[i]; ldsVT[d0 + i][r] = cv.hh; }
    }
    __syncthreads();

    // Scores: S(16x64) = Q(16x128) K^T. K fragment: lane = kv token, 8
    // contiguous d values per load -> straight global b128 loads.
    v8f sacc[4] = {};
#pragma unroll
    for (int d = 0; d < 4; ++d)
#pragma unroll
      for (int t = 0; t < 4; ++t) {
        Frag kb;
        const _Float16* kr =
            Kp + (size_t)(k0 + t * 16 + l16) * rs + d * 32 + hi * 8;
        kb.u[0] = *(const v4u*)(kr);
        kb.u[1] = *(const v4u*)(kr + 16);
        sacc[t] = wmma16x16x32(aq[d], kb, sacc[t]);
      }

    // Online softmax. Row e lives on one 16-lane half; xor masks 1/2/4/8
    // reduce across columns without crossing halves.
    const bool diag = (kt == qt);
#pragma unroll
    for (int e = 0; e < 8; ++e) {
      const int qrow = q0 + wave * 16 + hi * 8 + e;
      float sv[4];
      float pm = -1e30f;
#pragma unroll
      for (int t = 0; t < 4; ++t) {
        float v = sacc[t][e] * scale;
        if (diag && (k0 + t * 16 + l16) > qrow) v = -1e30f;
        sv[t] = v;
        pm = fmaxf(pm, v);
      }
#pragma unroll
      for (int m = 8; m >= 1; m >>= 1) pm = fmaxf(pm, __shfl_xor(pm, m, 32));
      const float mn = fmaxf(mrow[e], pm);
      float psum = 0.0f;
#pragma unroll
      for (int t = 0; t < 4; ++t) {
        float p = __expf(sv[t] - mn);
        psum += p;
        ldsP[wave * 16 + hi * 8 + e][t * 16 + l16] = (_Float16)p;
      }
#pragma unroll
      for (int m = 8; m >= 1; m >>= 1) psum += __shfl_xor(psum, m, 32);
      const float f = __expf(mrow[e] - mn);
      lrow[e] = lrow[e] * f + psum;
      mrow[e] = mn;
#pragma unroll
      for (int t = 0; t < 8; ++t) acc[t][e] = acc[t][e] * f;
    }
    __syncthreads();

    // O += P(16x64) V(64x128): P reloaded as A fragments, V^T as B fragments.
#pragma unroll
    for (int ks = 0; ks < 2; ++ks) {
      Frag ap;
      const _Float16* pp = &ldsP[wave * 16 + l16][ks * 32];
      ap.u[0] = *(const v4u*)(pp + hi * 8);
      ap.u[1] = *(const v4u*)(pp + 16 + hi * 8);
#pragma unroll
      for (int t = 0; t < 8; ++t) {
        Frag vb;
        const _Float16* vp2 = &ldsVT[t * 16 + l16][ks * 32 + hi * 8];
        vb.u[0] = *(const v4u*)(vp2);
        vb.u[1] = *(const v4u*)(vp2 + 16);
        acc[t] = wmma16x16x32(ap, vb, acc[t]);
      }
    }
    __syncthreads();
  }

  // Normalize and write attention output (f16, [token][h*Dh + d]).
#pragma unroll
  for (int e = 0; e < 8; ++e) {
    const float inv = 1.0f / lrow[e];
    const int row = q0 + wave * 16 + hi * 8 + e;
    _Float16* op = aout + (tok0 + row) * (size_t)C_ + (size_t)h * DH_;
#pragma unroll
    for (int t = 0; t < 8; ++t)
      op[t * 16 + l16] = (_Float16)(acc[t][e] * inv);
  }
}

// ---------------------------------------------------------------------------
// Host launcher
// ---------------------------------------------------------------------------
extern "C" void kernel_launch(void* const* d_in, const int* in_sizes, int n_in,
                              void* d_out, int out_size, void* d_ws,
                              size_t ws_size, hipStream_t stream) {
  (void)in_sizes; (void)n_in; (void)out_size; (void)ws_size;
  const float* x     = (const float*)d_in[0];
  const float* Wqkv  = (const float*)d_in[1];
  const float* Wproj = (const float*)d_in[2];
  float* out = (float*)d_out;

  const size_t nX     = (size_t)NTOK * C_;     //  8,388,608
  const size_t nWqkv  = (size_t)C_ * QKVC;     // 12,582,912
  const size_t nWproj = (size_t)C_ * C_;       //  4,194,304
  const size_t nQKV   = (size_t)NTOK * QKVC;   // 25,165,824

  _Float16* xh   = (_Float16*)d_ws;
  _Float16* wqh  = xh + nX;
  _Float16* wph  = wqh + nWqkv;
  _Float16* qkvh = wph + nWproj;
  _Float16* ah   = qkvh + nQKV;                // total ~112 MB of d_ws

  cast_f32_to_f16<<<(int)((nX + 255) / 256), 256, 0, stream>>>(x, xh, (int)nX);
  cast_f32_to_f16<<<(int)((nWqkv + 255) / 256), 256, 0, stream>>>(Wqkv, wqh,
                                                                  (int)nWqkv);
  cast_f32_to_f16<<<(int)((nWproj + 255) / 256), 256, 0, stream>>>(Wproj, wph,
                                                                   (int)nWproj);

  // qkv = x @ Wqkv  (4096 x 6144)
  gemm_f16_wmma<_Float16><<<dim3(QKVC / 256, NTOK / 128), 256, 0, stream>>>(
      xh, wqh, qkvh, NTOK, QKVC, C_);

  // RoPE on k and q thirds
  rope_inplace<<<(NTOK * 2 * H_ * 64) / 256, 256, 0, stream>>>(qkvh);

  // causal flash attention -> ah (4096 x 2048, f16)
  flash_attn_wmma<<<B_ * H_ * (S_ / 64), 128, 0, stream>>>(qkvh, ah);

  // out = ah @ Wproj (fp32 output)
  gemm_f16_wmma<float><<<dim3(C_ / 256, NTOK / 128), 256, 0, stream>>>(
      ah, wph, out, NTOK, C_, C_);
}
